// LSTM_19842748907626
// MI455X (gfx1250) — matrix operand order repaired and checked by reading, not compiled, not run on hardware
//
#include <hip/hip_runtime.h>

#define NCLASS 50257
#define EMB    256
#define HID    512
#define BATCH  256
#define TSTEPS 512
#define KTOT   (EMB + HID)          /* 768  */
#define NT_CLS ((NCLASS + 15) / 16) /* 3142 */

#define KHALF   384                 /* K staged per phase            */
#define BSTRIDE 392                 /* LDS row stride (elems), 16B-aligned, bank-staggered */

typedef __attribute__((ext_vector_type(8)))  __bf16 v8bf;
typedef __attribute__((ext_vector_type(16))) __bf16 v16bf;
typedef __attribute__((ext_vector_type(8)))  float  v8f;
typedef int v4i __attribute__((vector_size(16)));   // matches builtin param pointee

#define AS1 __attribute__((address_space(1)))
#define AS3 __attribute__((address_space(3)))

#if defined(__has_builtin)
#  if __has_builtin(__builtin_amdgcn_global_load_async_to_lds_b128)
#    define HAVE_ASYNC_LDS 1
#  else
#    define HAVE_ASYNC_LDS 0
#  endif
#  if __has_builtin(__builtin_amdgcn_s_wait_asynccnt)
#    define HAVE_WAIT_ASYNC 1
#  else
#    define HAVE_WAIT_ASYNC 0
#  endif
#else
#  define HAVE_ASYNC_LDS 0
#  define HAVE_WAIT_ASYNC 0
#endif

// ---------- helpers ----------
__device__ __forceinline__ unsigned short f2bf(float f) {
  unsigned int u = __builtin_bit_cast(unsigned int, f);
  u += 0x7FFFu + ((u >> 16) & 1u);            // round-to-nearest-even
  return (unsigned short)(u >> 16);
}

// A-matrix 16x32 bf16 tile per ISA 7.12.2: lanes 0-15 hold K {k0..+7, k0+16..+23},
// lanes 16-31 hold K {k0+8..+15, k0+24..+31}; M = lane&15 (row pointer per lane).
__device__ __forceinline__ v16bf load_a_tile(const unsigned short* row, int k0, int laneHi) {
  const v8bf lo = *(const v8bf*)(row + k0 + 8 * laneHi);
  const v8bf hi = *(const v8bf*)(row + k0 + 8 * laneHi + 16);
  return __builtin_shufflevector(lo, hi, 0,1,2,3,4,5,6,7,8,9,10,11,12,13,14,15);
}

// B-matrix 32x16 bf16 tile from a row pointer (global or LDS): lane = column N,
// lanes 0-15 K {k0..+7, k0+16..+23}, lanes 16-31 K {k0+8..+15, k0+24..+31}. Two 16B loads (16B-aligned).
__device__ __forceinline__ v16bf load_b_tile(const unsigned short* row, int k0, int laneHi) {
  const v8bf lo = *(const v8bf*)(row + k0 + 8 * laneHi);
  const v8bf hi = *(const v8bf*)(row + k0 + 8 * laneHi + 16);
  return __builtin_shufflevector(lo, hi, 0,1,2,3,4,5,6,7,8,9,10,11,12,13,14,15);
}

__device__ __forceinline__ v8f wmma_bf16(v16bf a, v16bf b, v8f c) {
  return __builtin_amdgcn_wmma_f32_16x16x32_bf16(false, a, false, b, (short)0, c,
                                                 false, false);
}

// 16B global -> LDS stage; async-to-LDS on CDNA5, register bounce otherwise.
__device__ __forceinline__ void stage16(const unsigned short* src, unsigned short* dst) {
#if HAVE_ASYNC_LDS
  __builtin_amdgcn_global_load_async_to_lds_b128((AS1 v4i*)src, (AS3 v4i*)dst, 0, 0);
#else
  *(v8bf*)dst = *(const v8bf*)src;
#endif
}

__device__ __forceinline__ void stage_wait() {
#if HAVE_ASYNC_LDS
#  if HAVE_WAIT_ASYNC
  __builtin_amdgcn_s_wait_asynccnt(0);
#  else
  asm volatile("s_wait_asynccnt 0x0" ::: "memory");
#  endif
#endif
}

__device__ __forceinline__ float sigm(float x) { return 1.0f / (1.0f + __expf(-x)); }

#define V8F_ZERO {0.f,0.f,0.f,0.f,0.f,0.f,0.f,0.f}

// ---------- one-time prep kernels ----------
__global__ __launch_bounds__(256) void f32_to_bf16_kernel(const float* __restrict__ src,
                                                          unsigned short* __restrict__ dst,
                                                          int n) {
  int i = blockIdx.x * 256 + threadIdx.x;
  if (i < n) dst[i] = f2bf(src[i]);
}

__global__ __launch_bounds__(256) void init_state_kernel(unsigned short* __restrict__ h_bf,
                                                         float* __restrict__ c_f32,
                                                         float* __restrict__ h_f32) {
  int i = blockIdx.x * 256 + threadIdx.x;  // BATCH*HID threads
  h_bf[i]  = 0x3F80;                       // bf16(1.0)
  c_f32[i] = 1.0f;
  h_f32[i] = 1.0f;
}

// ---------- fused LSTM step: g = [h|x] @ Wg^T + bg, gates, state update ----------
// One block (16 waves / 512 threads) per N-tile; the block's 4-gate B panel
// (64 Wg rows x 768 K) is staged K-half at a time into LDS via async copies,
// so Wg is fetched from L2 once per step instead of once per M-tile wave.
__global__ __launch_bounds__(512) void lstm_step_kernel(
    const unsigned short* __restrict__ h_in,   // [256,512] bf16
    unsigned short*       __restrict__ h_out,  // [256,512] bf16
    float*                __restrict__ c_st,   // [256,512] f32 (in-place)
    float*                __restrict__ h_f32,  // [256,512] f32
    const unsigned short* __restrict__ Wg,     // [2048,768] bf16 (x,i,f,o stacked)
    const unsigned short* __restrict__ Cemb,   // [NCLASS,256] bf16
    const int*            __restrict__ X,      // [256,512] tokens
    const float* __restrict__ bx, const float* __restrict__ bi,
    const float* __restrict__ bf_, const float* __restrict__ bo,
    int t)
{
  __shared__ unsigned short Bs[64 * BSTRIDE];  // ~50 KB

  const int tid    = threadIdx.x;
  const int lane   = tid & 31;
  const int laneHi = lane >> 4;
  const int l15    = lane & 15;
  const int tm     = tid >> 5;       // 16 M tiles (batch) = waves in block
  const int tn     = blockIdx.x;     // 32 N tiles (hidden)

  const int m  = tm * 16 + l15;      // batch row this lane loads (A)
  const int n  = tn * 16 + l15;      // gate-local column
  const int n0 = tn * 16;

  const unsigned short* hrow = h_in + m * HID;
  const int tok = X[m * TSTEPS + t];
  const unsigned short* erow = Cemb + (long)tok * EMB;
  __builtin_prefetch(erow, 0, 3);                  // global_prefetch_b8
  __builtin_prefetch(erow + 128, 0, 3);

  const unsigned short* BsX = Bs + (0 * 16 + l15) * BSTRIDE;
  const unsigned short* BsI = Bs + (1 * 16 + l15) * BSTRIDE;
  const unsigned short* BsF = Bs + (2 * 16 + l15) * BSTRIDE;
  const unsigned short* BsO = Bs + (3 * 16 + l15) * BSTRIDE;

  v8f ax = V8F_ZERO, ai = V8F_ZERO, af = V8F_ZERO, ao = V8F_ZERO;

  for (int kp = 0; kp < 2; ++kp) {
    // ---- stage this K-half of the B panel: 64 rows x 384 elems (16B x 3072) ----
    for (int i = tid; i < 3072; i += 512) {
      const int r   = i / 48;          // panel row 0..63  (gate*16 + col)
      const int off = (i % 48) * 8;    // elem offset within K-half
      const int g   = r >> 4;
      const int col = r & 15;
      stage16(Wg + (size_t)(g * HID + n0 + col) * KTOT + kp * KHALF + off,
              Bs + r * BSTRIDE + off);
    }
    stage_wait();
    __syncthreads();

    // ---- MACs over this K-half ----
#pragma unroll 4
    for (int kk = 0; kk < KHALF; kk += 32) {
      const int gk = kp * KHALF + kk;  // global K: [h | x] boundary at HID
      const unsigned short* ar = (gk < HID) ? (hrow + gk) : (erow + (gk - HID));
      const v16bf a = load_a_tile(ar, 0, laneHi);
      ax = wmma_bf16(a, load_b_tile(BsX, kk, laneHi), ax);
      ai = wmma_bf16(a, load_b_tile(BsI, kk, laneHi), ai);
      af = wmma_bf16(a, load_b_tile(BsF, kk, laneHi), af);
      ao = wmma_bf16(a, load_b_tile(BsO, kk, laneHi), ao);
    }
    __syncthreads();                   // protect LDS before next phase restages
  }

  // gate nonlinearities + state update (C/D layout: VGPR r -> M=r(+8 for hi lanes), N=l15)
  const float bxv = bx[n], biv = bi[n], bfv = bf_[n], bov = bo[n];
#pragma unroll
  for (int r = 0; r < 8; ++r) {
    const int  mg  = tm * 16 + r + laneHi * 8;
    const long idx = (long)mg * HID + n;
    const float gx = ax[r] + bxv;
    const float gi = ai[r] + biv;
    const float gf = af[r] + bfv;
    const float go = ao[r] + bov;
    const float alpha = tanhf(gx) * sigm(gi);
    const float cn    = c_st[idx] * sigm(gf) + alpha;
    const float hn    = tanhf(cn) + sigm(go);   // reference uses tanh(c)+sigmoid(o)
    c_st[idx]  = cn;
    h_f32[idx] = hn;
    h_out[idx] = f2bf(hn);
  }
}

// ---------- final classifier: out = h @ W^T + b ----------
__global__ __launch_bounds__(256) void lstm_cls_kernel(
    const unsigned short* __restrict__ hbf,   // [256,512] bf16
    const unsigned short* __restrict__ Wb,    // [NCLASS,512] bf16
    const float*          __restrict__ bias,  // [NCLASS]
    float*                __restrict__ out)   // [256,NCLASS]
{
  const int tid    = threadIdx.x;
  const int lane   = tid & 31;
  const int laneHi = lane >> 4;
  const int l15    = lane & 15;
  const int w      = blockIdx.x * 8 + (tid >> 5);  // 16*3142 waves
  const int tm     = w & 15;
  const int tn     = w >> 4;

  const int m = tm * 16 + l15;
  const int n = tn * 16 + l15;                     // class index
  const bool vok = (n < NCLASS);

  const unsigned short* hrow = hbf + m * HID;
  const unsigned short* Brow = Wb + (long)(vok ? n : 0) * HID;

  v8f acc = V8F_ZERO;
#pragma unroll 4
  for (int k0 = 0; k0 < HID; k0 += 32) {
    const v16bf a = load_a_tile(hrow, k0, laneHi);
    const v16bf b = load_b_tile(Brow, k0, laneHi);
    acc = wmma_bf16(a, b, acc);
  }

  const float bv = vok ? bias[n] : 0.0f;
#pragma unroll
  for (int r = 0; r < 8; ++r) {
    const int mg = tm * 16 + r + laneHi * 8;
    if (vok) out[(long)mg * NCLASS + n] = acc[r] + bv;
  }
}

__global__ __launch_bounds__(256) void copy_f32_kernel(const float* __restrict__ src,
                                                       float* __restrict__ dst, int n) {
  int i = blockIdx.x * 256 + threadIdx.x;
  if (i < n) dst[i] = src[i];
}

// ---------- host ----------
extern "C" void kernel_launch(void* const* d_in, const int* in_sizes, int n_in,
                              void* d_out, int out_size, void* d_ws, size_t ws_size,
                              hipStream_t stream) {
  (void)in_sizes; (void)n_in; (void)out_size; (void)ws_size;

  const int*   X      = (const int*)  d_in[0];
  const float* Cemb_f = (const float*)d_in[1];
  const float* Wxf    = (const float*)d_in[2];
  const float* bx     = (const float*)d_in[3];
  const float* Wif    = (const float*)d_in[4];
  const float* bi     = (const float*)d_in[5];
  const float* Wff    = (const float*)d_in[6];
  const float* bff    = (const float*)d_in[7];
  const float* Wof    = (const float*)d_in[8];
  const float* bo     = (const float*)d_in[9];
  const float* Wclsf  = (const float*)d_in[10];
  const float* bcls   = (const float*)d_in[11];
  float* out = (float*)d_out;

  // workspace carve-up (256B aligned)
  char*  ws  = (char*)d_ws;
  size_t off = 0;
  auto carve = [&](size_t bytes) {
    void* p = ws + off;
    off += (bytes + 255) & ~(size_t)255;
    return p;
  };
  unsigned short* Cemb_bf = (unsigned short*)carve((size_t)NCLASS * EMB * 2);
  unsigned short* Wg_bf   = (unsigned short*)carve((size_t)4 * HID * KTOT * 2);
  unsigned short* Wcls_bf = (unsigned short*)carve((size_t)NCLASS * HID * 2);
  unsigned short* hbf0    = (unsigned short*)carve((size_t)BATCH * HID * 2);
  unsigned short* hbf1    = (unsigned short*)carve((size_t)BATCH * HID * 2);
  float*          c_st    = (float*)carve((size_t)BATCH * HID * 4);
  float*          h_f32   = (float*)carve((size_t)BATCH * HID * 4);

  // 1) fp32 -> bf16 conversions (weights stay L2-resident afterwards)
  {
    int n = NCLASS * EMB;
    f32_to_bf16_kernel<<<(n + 255) / 256, 256, 0, stream>>>(Cemb_f, Cemb_bf, n);
    const float* gateSrc[4] = {Wxf, Wif, Wff, Wof};
    const int ng = HID * KTOT;  // 393216 per gate
    for (int g = 0; g < 4; ++g)
      f32_to_bf16_kernel<<<(ng + 255) / 256, 256, 0, stream>>>(
          gateSrc[g], Wg_bf + (size_t)g * ng, ng);
    int nc = NCLASS * HID;
    f32_to_bf16_kernel<<<(nc + 255) / 256, 256, 0, stream>>>(Wclsf, Wcls_bf, nc);
  }

  // 2) h0 = c0 = ones
  init_state_kernel<<<(BATCH * HID) / 256, 256, 0, stream>>>(hbf0, c_st, h_f32);

  // 3) recurrent scan: one fused GEMM+gates kernel per step, h ping-pong
  unsigned short* hin  = hbf0;
  unsigned short* hout = hbf1;
  for (int t = 0; t < TSTEPS; ++t) {
    lstm_step_kernel<<<32, 512, 0, stream>>>(hin, hout, c_st, h_f32, Wg_bf, Cemb_bf,
                                             X, bx, bi, bff, bo, t);
    unsigned short* tmp = hin; hin = hout; hout = tmp;
  }
  // after the swap, `hin` holds the final h (bf16)

  // 4) classifier GEMM  out[:, :NCLASS]
  lstm_cls_kernel<<<(16 * NT_CLS) / 8, 256, 0, stream>>>(hin, Wcls_bf, bcls, out);

  // 5) second output: final h in fp32
  copy_f32_kernel<<<(BATCH * HID) / 256, 256, 0, stream>>>(
      h_f32, out + (size_t)BATCH * NCLASS, BATCH * HID);
}
